// MetaBind_88974542504689
// MI455X (gfx1250) — matrix operand order repaired and checked by reading, not compiled, not run on hardware
//
#include <hip/hip_runtime.h>
#include <hip/hip_bf16.h>

// ---------------------------------------------------------------------------
// Problem constants (from reference)
// ---------------------------------------------------------------------------
#define NN 262144
#define DD 256
#define GG 1024
#define KK 4
#define HH 128
#define EPSV 0.01f

typedef __bf16 bf16_t;
typedef __bf16 v16bf __attribute__((ext_vector_type(16)));
typedef float  v8f   __attribute__((ext_vector_type(8)));

__device__ __forceinline__ float softplusf(float x) {
    // jax.nn.softplus = log1p(exp(x)), numerically stable
    return x > 20.f ? x : log1pf(expf(x));
}

__device__ __forceinline__ int lower_bound_i(const int* __restrict__ a, int n, int v) {
    int lo = 0, hi = n;
    while (lo < hi) {
        int mid = (lo + hi) >> 1;
        if (a[mid] < v) lo = mid + 1; else hi = mid;
    }
    return lo;
}

// ---------------------------------------------------------------------------
// 1) Pre-swizzle Wd1 (D x H, fp32 row-major) into WMMA B-fragment order, bf16.
//    For tile (ki, nt): B is 32(K) x 16(N).  Per ISA 7.12.2 (16-bit B):
//    lane l, element e:  K = (l>>4)*8 + (e&7) + (e>>3)*16,  N = l&15.
//    Packed so each lane's v16bf is contiguous: out[((ki*8+nt)*32 + l)*16 + e].
// ---------------------------------------------------------------------------
__global__ __launch_bounds__(256) void pack_wd1(const float* __restrict__ wd1,
                                                bf16_t* __restrict__ out) {
    int i = blockIdx.x * 256 + threadIdx.x;      // 0 .. 8*8*32*16-1 = 32767
    int e    = i & 15;
    int lane = (i >> 4) & 31;
    int tile = i >> 9;                            // ki*8 + nt
    int ki = tile >> 3, nt = tile & 7;
    int kk = ki * 32 + ((lane >> 4) << 3) + (e & 7) + ((e >> 3) << 4);    // global K (0..255)
    int nn = nt * 16 + (lane & 15);               // global N (0..127)
    out[i] = (bf16_t)wd1[kk * HH + nn];
}

// ---------------------------------------------------------------------------
// 2) Cluster constants.  h == 0 identically (segment_mean of centered values
//    with the same index), so rep = tanh(b1), broadcast over all G clusters.
// ---------------------------------------------------------------------------
__global__ __launch_bounds__(128) void cluster_const(const float* __restrict__ b1,
                                                     const float* __restrict__ Wm,
                                                     const float* __restrict__ bm,
                                                     const float* __restrict__ Wp,
                                                     const float* __restrict__ bp,
                                                     float* __restrict__ rep_ws,
                                                     float* __restrict__ prob_ws,
                                                     float* __restrict__ scale_ws) {
    __shared__ float srep[HH];
    int t = threadIdx.x;
    if (t < HH) {
        float v = tanhf(b1[t]);
        srep[t] = v;
        rep_ws[t] = v;
    }
    __syncthreads();
    if (t == 0) {
        float lg[KK];
        #pragma unroll
        for (int k = 0; k < KK; ++k) lg[k] = bm[k];
        for (int h = 0; h < HH; ++h) {
            float rv = srep[h];
            #pragma unroll
            for (int k = 0; k < KK; ++k) lg[k] += rv * Wm[h * KK + k];
        }
        float mx = lg[0];
        #pragma unroll
        for (int k = 1; k < KK; ++k) mx = fmaxf(mx, lg[k]);
        float s = 0.f, ev[KK];
        #pragma unroll
        for (int k = 0; k < KK; ++k) { ev[k] = expf(lg[k] - mx); s += ev[k]; }
        #pragma unroll
        for (int k = 0; k < KK; ++k) prob_ws[k] = ev[k] / s;
        for (int k = 0; k < KK; ++k) {
            float pr = bp[k];
            for (int h = 0; h < HH; ++h) pr += srep[h] * Wp[h * (2 * KK) + k];
            scale_ws[k] = softplusf(pr) * (1.f - EPSV) + EPSV;
        }
    }
}

// ---------------------------------------------------------------------------
// 3) Segment means of target_x / target_y.  target_index is sorted, so each
//    block binary-searches its segment bounds -> deterministic, atomic-free.
// ---------------------------------------------------------------------------
__global__ __launch_bounds__(256) void seg_mean(const float* __restrict__ x,
                                                const float* __restrict__ y,
                                                const int* __restrict__ idx,
                                                float* __restrict__ xa,
                                                float* __restrict__ ya) {
    int g  = blockIdx.x;
    int lo = lower_bound_i(idx, NN, g);
    int hi = lower_bound_i(idx, NN, g + 1);
    int t  = threadIdx.x;                       // column 0..255

    float acc = 0.f;
    int r = lo;
    for (; r + 4 <= hi; r += 4) {
        float v0 = x[(size_t)(r + 0) * DD + t];
        float v1 = x[(size_t)(r + 1) * DD + t];
        float v2 = x[(size_t)(r + 2) * DD + t];
        float v3 = x[(size_t)(r + 3) * DD + t];
        acc += v0; acc += v1; acc += v2; acc += v3;
    }
    for (; r < hi; ++r) acc += x[(size_t)r * DD + t];

    float cnt = (float)((hi - lo) > 1 ? (hi - lo) : 1);
    xa[(size_t)g * DD + t] = acc / cnt;

    // y: block reduction (deterministic order)
    __shared__ float red[256];
    float ay = 0.f;
    for (int rr = lo + t; rr < hi; rr += 256) ay += y[rr];
    red[t] = ay;
    __syncthreads();
    for (int s = 128; s > 0; s >>= 1) {
        if (t < s) red[t] += red[t + s];
        __syncthreads();
    }
    if (t == 0) ya[g] = red[0] / cnt;
}

// ---------------------------------------------------------------------------
// 4) Broadcast the constant cluster outputs into d_out.
//    d_out layout: [y_pred N][y_sigma_sum N][t_prob G*K][t_rep G*H][dist_kl G][t_scale G*K]
// ---------------------------------------------------------------------------
__global__ __launch_bounds__(256) void broadcast_out(const float* __restrict__ rep,
                                                     const float* __restrict__ prob,
                                                     const float* __restrict__ scale,
                                                     float* __restrict__ out) {
    const size_t o2 = 2 * (size_t)NN;
    const size_t o3 = o2 + (size_t)GG * KK;
    const size_t o4 = o3 + (size_t)GG * HH;
    const size_t o5 = o4 + (size_t)GG;
    int i = blockIdx.x * 256 + threadIdx.x;
    if (i < GG * HH) out[o3 + i] = rep[i & (HH - 1)];
    if (i < GG * KK) {
        out[o2 + i] = prob[i & (KK - 1)];
        out[o5 + i] = scale[i & (KK - 1)];
    }
    if (i < GG) out[o4 + i] = 0.f;   // dist_kl == 0 identically (c_* == t_*)
}

// ---------------------------------------------------------------------------
// 5) Decoder: pred = tanh((x - txa[idx]) @ Wd1 + bd1) @ Wd2 + bd2, then the
//    mixture head.  8 waves/block, 16 rows per wave, WMMA bf16 16x16x32.
// ---------------------------------------------------------------------------
__global__ __launch_bounds__(256) void decoder(const float* __restrict__ tx,
                                               const int* __restrict__ tidx,
                                               const float* __restrict__ txa,
                                               const float* __restrict__ tya,
                                               const bf16_t* __restrict__ wd1p,
                                               const float* __restrict__ bd1,
                                               const float* __restrict__ wd2,
                                               const float* __restrict__ bd2,
                                               const float* __restrict__ prob,
                                               const float* __restrict__ scale,
                                               float* __restrict__ ypred,
                                               float* __restrict__ ysig) {
    __shared__ float s_wd2[HH * 2 * KK];   // 1024
    __shared__ float s_bd1[HH];
    __shared__ float s_bd2[2 * KK];
    __shared__ float s_ps[2 * KK];         // [0..3]=prob, [4..7]=scale

    int t = threadIdx.x;
    for (int i = t; i < HH * 2 * KK; i += 256) s_wd2[i] = wd2[i];
    if (t < HH) s_bd1[t] = bd1[t];
    if (t < 2 * KK) s_bd2[t] = bd2[t];
    if (t < KK) { s_ps[t] = prob[t]; s_ps[KK + t] = scale[t]; }
    __syncthreads();

    int wave = t >> 5;
    int lane = t & 31;
    int row0 = blockIdx.x * 128 + wave * 16;
    int m    = lane & 15;                  // row within strip (A layout), also N within B/C tiles
    int kb   = (lane >> 4) << 3;           // 0 or 8: K sub-offset for A fragment
    int arow = row0 + m;
    int seg  = tidx[arow];

    const float* xr = tx  + (size_t)arow * DD;
    const float* ar = txa + (size_t)seg  * DD;

    v8f acc[8];
    #pragma unroll
    for (int nt = 0; nt < 8; ++nt) acc[nt] = (v8f){0.f,0.f,0.f,0.f,0.f,0.f,0.f,0.f};

    #pragma unroll
    for (int ki = 0; ki < 8; ++ki) {
        int k0 = ki * 32 + kb;
        // A fragment: element e -> K = k0 + e (e<8) else k0 + 16 + (e-8)
        float4 x0 = *(const float4*)(xr + k0);
        float4 x1 = *(const float4*)(xr + k0 + 4);
        float4 x2 = *(const float4*)(xr + k0 + 16);
        float4 x3 = *(const float4*)(xr + k0 + 20);
        float4 a0 = *(const float4*)(ar + k0);
        float4 a1 = *(const float4*)(ar + k0 + 4);
        float4 a2 = *(const float4*)(ar + k0 + 16);
        float4 a3 = *(const float4*)(ar + k0 + 20);
        v16bf a;
        a[0]  = (bf16_t)(x0.x - a0.x);  a[1]  = (bf16_t)(x0.y - a0.y);
        a[2]  = (bf16_t)(x0.z - a0.z);  a[3]  = (bf16_t)(x0.w - a0.w);
        a[4]  = (bf16_t)(x1.x - a1.x);  a[5]  = (bf16_t)(x1.y - a1.y);
        a[6]  = (bf16_t)(x1.z - a1.z);  a[7]  = (bf16_t)(x1.w - a1.w);
        a[8]  = (bf16_t)(x2.x - a2.x);  a[9]  = (bf16_t)(x2.y - a2.y);
        a[10] = (bf16_t)(x2.z - a2.z);  a[11] = (bf16_t)(x2.w - a2.w);
        a[12] = (bf16_t)(x3.x - a3.x);  a[13] = (bf16_t)(x3.y - a3.y);
        a[14] = (bf16_t)(x3.z - a3.z);  a[15] = (bf16_t)(x3.w - a3.w);

        #pragma unroll
        for (int nt = 0; nt < 8; ++nt) {
            v16bf b = *(const v16bf*)(wd1p + (((size_t)(ki * 8 + nt) * 32 + lane) << 4));
            acc[nt] = __builtin_amdgcn_wmma_f32_16x16x32_bf16(
                false, a, false, b, (short)0, acc[nt], false, false);
        }
    }

    // bias + tanh in C layout: element r of acc[nt] is row (lane<16 ? r : r+8),
    // column nt*16 + m.
    #pragma unroll
    for (int nt = 0; nt < 8; ++nt) {
        float bv = s_bd1[nt * 16 + m];
        #pragma unroll
        for (int r = 0; r < 8; ++r) acc[nt][r] = tanhf(acc[nt][r] + bv);
    }

    // Second GEMM (h1 @ Wd2): per-lane partials over its 8 columns, then
    // butterfly reduce across the 16 lanes that share the same 8 rows.
    float part[8][8];
    #pragma unroll
    for (int r = 0; r < 8; ++r)
        #pragma unroll
        for (int j = 0; j < 8; ++j) part[r][j] = 0.f;

    #pragma unroll
    for (int nt = 0; nt < 8; ++nt) {
        float w2v[8];
        #pragma unroll
        for (int j = 0; j < 8; ++j) w2v[j] = s_wd2[(nt * 16 + m) * 8 + j];
        #pragma unroll
        for (int r = 0; r < 8; ++r) {
            float hv = acc[nt][r];
            #pragma unroll
            for (int j = 0; j < 8; ++j) part[r][j] += hv * w2v[j];
        }
    }

    #pragma unroll
    for (int off = 1; off < 16; off <<= 1) {
        #pragma unroll
        for (int r = 0; r < 8; ++r)
            #pragma unroll
            for (int j = 0; j < 8; ++j)
                part[r][j] += __shfl_xor(part[r][j], off, 32);
    }

    // lanes 0-7 own rows 0-7; lanes 16-23 own rows 8-15
    if ((lane & 8) == 0) {
        int r = lane & 7;
        int rowout = ((lane >> 4) << 3) + r;
        int grow = row0 + rowout;
        float pred[8];
        #pragma unroll
        for (int j = 0; j < 8; ++j) pred[j] = part[r][j] + s_bd2[j];

        float yp = 0.f, ss = 0.f;
        #pragma unroll
        for (int k = 0; k < KK; ++k) {
            float pk  = s_ps[k];
            float sck = s_ps[KK + k];
            float mean = pred[k];
            float sig  = softplusf(pred[KK + k]) * (1.f - EPSV) + EPSV;
            yp += pk * (sck * mean);
            float yk = sck * pk * sig;
            ss += yk * yk;
        }
        int sg = tidx[grow];
        yp += tya[sg];
        ypred[grow] = yp;
        ysig[grow]  = sqrtf(ss);
    }
}

// ---------------------------------------------------------------------------
// Launch
// ---------------------------------------------------------------------------
extern "C" void kernel_launch(void* const* d_in, const int* in_sizes, int n_in,
                              void* d_out, int out_size, void* d_ws, size_t ws_size,
                              hipStream_t stream) {
    (void)in_sizes; (void)n_in; (void)out_size; (void)ws_size;

    // setup_inputs order
    const int*   tidx = (const int*)d_in[3];
    const float* tx   = (const float*)d_in[4];
    const float* ty   = (const float*)d_in[5];
    const float* b1   = (const float*)d_in[7];
    const float* Wm   = (const float*)d_in[8];
    const float* bm   = (const float*)d_in[9];
    const float* Wp   = (const float*)d_in[10];
    const float* bp   = (const float*)d_in[11];
    const float* Wd1  = (const float*)d_in[12];
    const float* bd1  = (const float*)d_in[13];
    const float* Wd2  = (const float*)d_in[14];
    const float* bd2  = (const float*)d_in[15];
    // context_* and W1 provably do not affect the outputs (h == 0 identically).

    char* ws = (char*)d_ws;
    float*  txa   = (float*)(ws + 0);            // G*D*4      = 1048576 B
    float*  tya   = (float*)(ws + 1048576);      // G*4        = 4096 B
    float*  rep   = (float*)(ws + 1052672);      // H*4        = 512 B
    float*  prob  = (float*)(ws + 1053184);      // 64 B
    float*  scale = (float*)(ws + 1053248);      // 64 B
    bf16_t* wd1p  = (bf16_t*)(ws + 1053312);     // 32768*2    = 65536 B  (32B aligned)

    float* out = (float*)d_out;

    pack_wd1<<<(8 * 8 * 32 * 16) / 256, 256, 0, stream>>>(Wd1, wd1p);
    cluster_const<<<1, 128, 0, stream>>>(b1, Wm, bm, Wp, bp, rep, prob, scale);
    seg_mean<<<GG, 256, 0, stream>>>(tx, ty, tidx, txa, tya);
    broadcast_out<<<(GG * HH + 255) / 256, 256, 0, stream>>>(rep, prob, scale, out);
    decoder<<<NN / 128, 256, 0, stream>>>(tx, tidx, txa, tya, wd1p,
                                          bd1, Wd2, bd2, prob, scale,
                                          out /*y_pred*/, out + NN /*y_sigma_sum*/);
}
